// MHA_42614665511290
// MI455X (gfx1250) — matrix operand order repaired and checked by reading, not compiled, not run on hardware
//
#include <hip/hip_runtime.h>
#include <hip/hip_bf16.h>

#define DEVINL static __device__ __forceinline__

typedef __attribute__((ext_vector_type(16))) __bf16 v16bf;
typedef __attribute__((ext_vector_type(8)))  __bf16 v8bf;
typedef __attribute__((ext_vector_type(8)))  float  v8f;
typedef __attribute__((ext_vector_type(4)))  unsigned u32x4;
typedef __attribute__((ext_vector_type(8)))  int      i32x8;
typedef __attribute__((ext_vector_type(4)))  int      i32x4;

constexpr int Bx = 2, Sx = 2048, Ex = 2048, Hx = 16, Dx = 128;
constexpr int TOK = Bx * Sx;
constexpr size_t MB = 1024ull * 1024ull;

#if __has_builtin(__builtin_amdgcn_tensor_load_to_lds) && \
    __has_builtin(__builtin_amdgcn_s_wait_tensorcnt)
#define HAVE_TDM 1
#else
#define HAVE_TDM 0
#endif

// Compiler-visible "LDS may have been written" point (TDM writes are invisible
// to alias analysis; without this the staged-tile loads fold to undef and hoist).
#define LDS_PUBLISH() asm volatile("" ::: "memory")

// ---------- helpers ----------

DEVINL __bf16 f2bf(float f) {
  unsigned u = __builtin_bit_cast(unsigned, f);
  unsigned r = u + 0x7FFFu + ((u >> 16) & 1u);   // round-to-nearest-even
  unsigned short s = (unsigned short)(r >> 16);
  return __builtin_bit_cast(__bf16, s);
}

DEVINL v8f wmma_bf16(v16bf a, v16bf b, v8f c) {
  return __builtin_amdgcn_wmma_f32_16x16x32_bf16(false, a, false, b, (short)0, c,
                                                 false, false);
}

// A-fragment: 16x32 bf16, M-rows contiguous in memory.
DEVINL v16bf load_fragA(const __bf16* base, int row0, int ld, int k0, int lane) {
  int r = lane & 15, half = lane >> 4;
  const __bf16* p = base + (size_t)(row0 + r) * ld + k0 + half * 8;
  v8bf lo = *(const v8bf*)p;
  v8bf hi = *(const v8bf*)(p + 16);
  v16bf f;
#pragma unroll
  for (int i = 0; i < 8; ++i) { f[i] = lo[i]; f[i + 8] = hi[i]; }
  return f;
}

// B-fragment: 32x16 where B[k][n] = M[n][k]; lane n holds k = 8h..8h+7 and 16+8h..16+8h+7.
DEVINL v16bf load_fragB(const __bf16* base, int n0, int ld, int k0, int lane) {
  int n = lane & 15, half = lane >> 4;
  const __bf16* p = base + (size_t)(n0 + n) * ld + k0 + half * 8;
  v8bf lo = *(const v8bf*)p;
  v8bf hi = *(const v8bf*)(p + 16);
  v16bf f;
#pragma unroll
  for (int i = 0; i < 8; ++i) { f[i] = lo[i]; f[i + 8] = hi[i]; }
  return f;
}

#if HAVE_TDM
// One TDM descriptor moves a [rows x 32] bf16 tile (row stride = K elements)
// from global into LDS at byte offset lds_off. 2-D tensor, data_size = 2B.
DEVINL void tdm_stage(unsigned lds_off, const __bf16* gsrc, int K, int rows) {
  unsigned long long ga = (unsigned long long)(size_t)gsrc;
  u32x4 g0;
  g0[0] = 1u;                                            // count=1 (user D#)
  g0[1] = lds_off;                                       // lds_addr (bytes)
  g0[2] = (unsigned)(ga & 0xffffffffu);                  // global_addr[31:0]
  g0[3] = (unsigned)((ga >> 32) & 0x01ffffffu) | (2u << 30); // addr[56:32] | type=2
  i32x8 g1;
  g1[0] = 0x00010000;                   // wg_mask=0, data_size=1 (2 bytes)
  g1[1] = (int)(32u << 16);             // tensor_dim0[15:0] = 32 (tile k width)
  g1[2] = (int)((unsigned)rows << 16);  // tensor_dim0 hi=0 | tensor_dim1 lo = rows
  g1[3] = (int)(32u << 16);             // tensor_dim1 hi=0 | tile_dim0 = 32
  g1[4] = rows;                         // tile_dim1 = rows, tile_dim2 = 0
  g1[5] = K;                            // tensor_dim0_stride[31:0] = K elements
  g1[6] = 0;
  g1[7] = 0;
  i32x4 z4 = {};
#if defined(__clang_major__) && __clang_major__ >= 23
  i32x8 z8 = {};
  __builtin_amdgcn_tensor_load_to_lds(g0, g1, z4, z4, z8, 0);
#else
  __builtin_amdgcn_tensor_load_to_lds(g0, g1, z4, z4, 0);
#endif
}
#endif

// ---------- kernels ----------

__global__ __launch_bounds__(256) void cvt_f32_bf16(const float* __restrict__ in,
                                                    __bf16* __restrict__ out, size_t n) {
  size_t i = (size_t)blockIdx.x * blockDim.x + threadIdx.x;
  if (i < n) out[i] = f2bf(in[i]);
}

// out[m,n] = sum_k A[m,k]*W[n,k] + bias[n]; A: MxK bf16, W: NxK bf16 (row-major).
// Block: 256 thr = 8 waves; tile 128(M) x 128(N); W tile staged in LDS (double
// buffer, via TDM when available). A-fragments are register-prefetched.
template <int QKV>
__global__ __launch_bounds__(256) void gemm_wmma(const __bf16* __restrict__ A,
                                                 const __bf16* __restrict__ W,
                                                 const float* __restrict__ bias,
                                                 float* __restrict__ out, int K) {
  // NOTE: only __shared__ object in this kernel => LDS offset 0 (used by TDM D#).
  __shared__ __align__(16) __bf16 wtile[2][128 * 32];
  int tid = threadIdx.x;
  int lane = tid & 31, wave = tid >> 5;
  int row0 = blockIdx.x * 128 + wave * 16;
  int col0 = blockIdx.y * 128;

#if HAVE_TDM
  if (wave == 0) {
    tdm_stage(0u, W + (size_t)col0 * K, K, 128);
    __builtin_amdgcn_s_wait_tensorcnt(0);
  }
#else
#pragma unroll
  for (int i = 0; i < 2; ++i) {
    int c = tid + 256 * i, rr = c >> 2, o = (c & 3) * 8;
    *(v8bf*)(&wtile[0][0] + rr * 32 + o) =
        *(const v8bf*)(W + (size_t)(col0 + rr) * K + o);
  }
#endif
  __syncthreads();
  LDS_PUBLISH();

  v8f acc[8] = {};
  v16bf a = load_fragA(A, row0, K, 0, lane);
  for (int k0 = 0; k0 < K; k0 += 32) {
    int cur = (k0 >> 5) & 1;
    bool more = (k0 + 32) < K;
    if (more) {
#if HAVE_TDM
      if (wave == 0)
        tdm_stage((unsigned)((cur ^ 1) * 128 * 32 * 2),
                  W + (size_t)col0 * K + (k0 + 32), K, 128);
#else
#pragma unroll
      for (int i = 0; i < 2; ++i) {
        int c = tid + 256 * i, rr = c >> 2, o = (c & 3) * 8;
        *(v8bf*)(&wtile[cur ^ 1][0] + rr * 32 + o) =
            *(const v8bf*)(W + (size_t)(col0 + rr) * K + (k0 + 32) + o);
      }
#endif
    }
    // prefetch next A fragment (clamped address on last iter; value unused then)
    v16bf an = load_fragA(A, row0, K, more ? (k0 + 32) : 0, lane);
    v16bf b[8];
#pragma unroll
    for (int j = 0; j < 8; ++j) b[j] = load_fragB(&wtile[cur][0], 16 * j, 32, 0, lane);
#pragma unroll
    for (int j = 0; j < 8; ++j) acc[j] = wmma_bf16(a, b[j], acc[j]);
    a = an;
    if (more) {
#if HAVE_TDM
      if (wave == 0) __builtin_amdgcn_s_wait_tensorcnt(0);
#endif
      __syncthreads();
      LDS_PUBLISH();
    }
  }

  int n16 = lane & 15, hi = lane >> 4;
#pragma unroll
  for (int j = 0; j < 8; ++j) {
    int n = col0 + 16 * j + n16;
    float bv = bias[n];
#pragma unroll
    for (int i = 0; i < 8; ++i) {
      int m = row0 + i + 8 * hi;
      float v = acc[j][i] + bv;
      if (QKV) {
        int b2 = m >> 11, s = m & (Sx - 1), h = n >> 7, d = n & (Dx - 1);
        out[(((size_t)(b2 * Hx + h)) * Sx + s) * Dx + d] = v;
      } else {
        out[(size_t)m * Ex + n] = v;
      }
    }
  }
}

// RoPE on f32 [BH,S,D] q/k -> bf16 [BH,S,D]; one thread per (bh,s,d<64).
__global__ __launch_bounds__(256) void rope_cvt(const float* __restrict__ qf,
                                                const float* __restrict__ kf,
                                                __bf16* __restrict__ qb,
                                                __bf16* __restrict__ kb) {
  size_t idx = (size_t)blockIdx.x * blockDim.x + threadIdx.x;
  int d = (int)(idx & 63);
  size_t t = idx >> 6;
  int s = (int)(t & (size_t)(Sx - 1));
  size_t bh = t >> 11;
  size_t base = bh * (size_t)(Sx * Dx) + (size_t)s * Dx;
  float inv = __expf(-(float)d * (9.2103403719761836f / 64.f));
  float ang = (float)s * inv;
  float sn, cs;
  __sincosf(ang, &sn, &cs);
  float q1 = qf[base + d], q2 = qf[base + d + 64];
  qb[base + d]      = f2bf(q1 * cs - q2 * sn);
  qb[base + d + 64] = f2bf(q1 * sn + q2 * cs);
  float k1 = kf[base + d], k2 = kf[base + d + 64];
  kb[base + d]      = f2bf(k1 * cs - k2 * sn);
  kb[base + d + 64] = f2bf(k1 * sn + k2 * cs);
}

// V (f32 [BH,S,D]) -> V^T (bf16 [BH,D,S]) via 16x16 LDS tiles.
__global__ __launch_bounds__(256) void transpose_v(const float* __restrict__ vf,
                                                   __bf16* __restrict__ vt) {
  __shared__ float tile[16][17];
  int c = threadIdx.x & 15, r = threadIdx.x >> 4;
  size_t bh = blockIdx.z;
  int s0 = blockIdx.x * 16, d0 = blockIdx.y * 16;
  tile[r][c] = vf[bh * (size_t)(Sx * Dx) + (size_t)(s0 + r) * Dx + d0 + c];
  __syncthreads();
  vt[bh * (size_t)(Dx * Sx) + (size_t)(d0 + r) * Sx + s0 + c] = f2bf(tile[c][r]);
}

// Causal flash attention: 4 waves/block, one 16-row q-tile per wave.
__global__ __launch_bounds__(128) void flash_attn(const __bf16* __restrict__ Q,
                                                  const __bf16* __restrict__ Kb,
                                                  const __bf16* __restrict__ Vt,
                                                  __bf16* __restrict__ attn) {
  __shared__ __align__(16) __bf16 pls[4][16 * 32];
  int lane = threadIdx.x & 31, wave = threadIdx.x >> 5;
  int bh = blockIdx.z * Hx + blockIdx.y;
  int q0 = (blockIdx.x * 4 + wave) * 16;
  const __bf16* Qp = Q  + (size_t)bh * Sx * Dx;
  const __bf16* Kp = Kb + (size_t)bh * Sx * Dx;
  const __bf16* Vp = Vt + (size_t)bh * Dx * Sx;

  v16bf qfr[4];
#pragma unroll
  for (int kk = 0; kk < 4; ++kk) qfr[kk] = load_fragA(Qp, q0, Dx, kk * 32, lane);

  v8f acc[8] = {};
  float mrun[8], lrun[8];
#pragma unroll
  for (int i = 0; i < 8; ++i) { mrun[i] = -3.0e38f; lrun[i] = 0.f; }

  int hi = lane >> 4, n16 = lane & 15;
  const float sc = 0.08838834764831845f;  // 1/sqrt(128)
  int nblocks = (q0 + 47) / 32;

  for (int t0 = 0; t0 < nblocks * 32; t0 += 32) {
    // all K fragments first, then the 8 score WMMAs back-to-back
    v16bf kb0[4], kb1[4];
#pragma unroll
    for (int kk = 0; kk < 4; ++kk) {
      kb0[kk] = load_fragB(Kp, t0,      Dx, kk * 32, lane);
      kb1[kk] = load_fragB(Kp, t0 + 16, Dx, kk * 32, lane);
    }
    v8f s0 = {}, s1 = {};
#pragma unroll
    for (int kk = 0; kk < 4; ++kk) s0 = wmma_bf16(qfr[kk], kb0[kk], s0);
#pragma unroll
    for (int kk = 0; kk < 4; ++kk) s1 = wmma_bf16(qfr[kk], kb1[kk], s1);

    // prefetch V fragments now; they are independent of the softmax below
    v16bf vb[8];
#pragma unroll
    for (int j = 0; j < 8; ++j) vb[j] = load_fragB(Vp, j * 16, Sx, t0, lane);

    bool edge = (t0 + 31) > q0;
    float corr[8];
#pragma unroll
    for (int i = 0; i < 8; ++i) {
      float a = s0[i] * sc, b = s1[i] * sc;
      if (edge) {
        int qg = q0 + i + 8 * hi;
        if (t0 + n16 > qg)      a = -3.0e38f;
        if (t0 + 16 + n16 > qg) b = -3.0e38f;
      }
      float v = fmaxf(a, b);
#pragma unroll
      for (int m = 1; m < 16; m <<= 1) v = fmaxf(v, __shfl_xor(v, m, 32));
      float mnew = fmaxf(mrun[i], v);
      float p0 = __expf(a - mnew);
      float p1 = __expf(b - mnew);
      float rs = p0 + p1;
#pragma unroll
      for (int m = 1; m < 16; m <<= 1) rs += __shfl_xor(rs, m, 32);
      float cr = __expf(mrun[i] - mnew);
      lrun[i] = lrun[i] * cr + rs;
      mrun[i] = mnew;
      corr[i] = cr;
      s0[i] = p0; s1[i] = p1;
    }
#pragma unroll
    for (int j = 0; j < 8; ++j)
#pragma unroll
      for (int i = 0; i < 8; ++i) acc[j][i] *= corr[i];

    // C-layout probs -> LDS 16x32 -> reload in A-fragment layout
    __bf16* pb = &pls[wave][0];
#pragma unroll
    for (int i = 0; i < 8; ++i) {
      int row = i + 8 * hi;
      pb[row * 32 + n16]      = f2bf(s0[i]);
      pb[row * 32 + n16 + 16] = f2bf(s1[i]);
    }
    asm volatile("s_wait_dscnt 0" ::: "memory");
    v16bf pfrag = load_fragA(pb, 0, 32, 0, lane);
#pragma unroll
    for (int j = 0; j < 8; ++j) acc[j] = wmma_bf16(pfrag, vb[j], acc[j]);
  }

  int bb = blockIdx.z, h = blockIdx.y;
#pragma unroll
  for (int i = 0; i < 8; ++i) {
    int row = i + 8 * hi;
    float rl = 1.f / lrun[i];
    size_t rowoff = ((size_t)(bb * Sx + q0 + row)) * Ex + (size_t)h * Dx + n16;
#pragma unroll
    for (int j = 0; j < 8; ++j)
      attn[rowoff + j * 16] = f2bf(acc[j][i] * rl);
  }
}

// ---------- launch ----------

extern "C" void kernel_launch(void* const* d_in, const int* in_sizes, int n_in,
                              void* d_out, int out_size, void* d_ws, size_t ws_size,
                              hipStream_t stream) {
  const float* x  = (const float*)d_in[0];
  const float* wq = (const float*)d_in[1];
  const float* bq = (const float*)d_in[2];
  const float* wk = (const float*)d_in[3];
  const float* bk = (const float*)d_in[4];
  const float* wv = (const float*)d_in[5];
  const float* bv = (const float*)d_in[6];
  const float* wo = (const float*)d_in[7];
  const float* bo = (const float*)d_in[8];
  float* out = (float*)d_out;

  char* ws = (char*)d_ws;
  __bf16* xbf = (__bf16*)(ws + 0 * MB);
  __bf16* wqb = (__bf16*)(ws + 16 * MB);
  __bf16* wkb = (__bf16*)(ws + 24 * MB);
  __bf16* wvb = (__bf16*)(ws + 32 * MB);
  __bf16* wob = (__bf16*)(ws + 40 * MB);
  float*  qf  = (float*)(ws + 48 * MB);
  float*  kf  = (float*)(ws + 80 * MB);
  float*  vf  = (float*)(ws + 112 * MB);
  __bf16* qb  = (__bf16*)(ws + 144 * MB);
  __bf16* kb2 = (__bf16*)(ws + 160 * MB);
  __bf16* vt  = (__bf16*)(ws + 176 * MB);
  __bf16* at  = (__bf16*)(ws + 0 * MB);   // reuse x-bf16 region after projections

  size_t nx = (size_t)TOK * Ex, nw = (size_t)Ex * Ex;
  cvt_f32_bf16<<<dim3((nx + 255) / 256), 256, 0, stream>>>(x,  xbf, nx);
  cvt_f32_bf16<<<dim3((nw + 255) / 256), 256, 0, stream>>>(wq, wqb, nw);
  cvt_f32_bf16<<<dim3((nw + 255) / 256), 256, 0, stream>>>(wk, wkb, nw);
  cvt_f32_bf16<<<dim3((nw + 255) / 256), 256, 0, stream>>>(wv, wvb, nw);
  cvt_f32_bf16<<<dim3((nw + 255) / 256), 256, 0, stream>>>(wo, wob, nw);

  dim3 ggrid(TOK / 128, Ex / 128);
  gemm_wmma<1><<<ggrid, 256, 0, stream>>>(xbf, wqb, bq, qf, Ex);
  gemm_wmma<1><<<ggrid, 256, 0, stream>>>(xbf, wkb, bk, kf, Ex);
  gemm_wmma<1><<<ggrid, 256, 0, stream>>>(xbf, wvb, bv, vf, Ex);

  size_t nrope = (size_t)Bx * Hx * Sx * (Dx / 2);
  rope_cvt<<<dim3((nrope + 255) / 256), 256, 0, stream>>>(qf, kf, qb, kb2);
  transpose_v<<<dim3(Sx / 16, Dx / 16, Bx * Hx), 256, 0, stream>>>(vf, vt);

  flash_attn<<<dim3(Sx / 64, Hx, Bx), 128, 0, stream>>>(qb, kb2, vt, at);

  gemm_wmma<0><<<ggrid, 256, 0, stream>>>(at, wob, bo, out, Ex);
}
